// GroupedExperts_11991548691208
// MI455X (gfx1250) — compile-verified, matrix operand order checked by reading, not run on hardware
//
#include <hip/hip_runtime.h>

typedef __attribute__((ext_vector_type(16))) __bf16 v16bf;
typedef __attribute__((ext_vector_type(8)))  __bf16 v8bf;
typedef __attribute__((ext_vector_type(8)))  float  v8f;
typedef __attribute__((ext_vector_type(4)))  unsigned int u32x4;
typedef __attribute__((ext_vector_type(2)))  unsigned int u32x2;
typedef __attribute__((ext_vector_type(4)))  int i32x4;

#define NE 8
#define NT 1024
#define ND 1024
#define NH 2048

// GEMM tiling (shared by both GEMM kernels)
#define BM 64
#define BN 128
#define BK 64
#define NTHREADS 256   // 8 waves; waves arranged 2 (M) x 4 (N), wave tile 32x32

// Use CDNA5 async global->LDS DMA when the toolchain exposes it.
#if defined(__has_builtin)
#if __has_builtin(__builtin_amdgcn_global_load_async_to_lds_b128) && \
    __has_builtin(__builtin_amdgcn_s_wait_asynccnt)
#define USE_ASYNC 1
#endif
#endif
#ifndef USE_ASYNC
#define USE_ASYNC 0
#endif

// fp32 -> bf16 round-to-nearest-even (bit pattern)
static __device__ __forceinline__ unsigned short f2bf(float f) {
  unsigned u = __float_as_uint(f);
  unsigned r = u + 0x7FFFu + ((u >> 16) & 1u);
  return (unsigned short)(r >> 16);
}

// 16-byte global -> LDS copy (async DMA when available).
static __device__ __forceinline__ void cp16_g2l(unsigned short* l,
                                                const unsigned short* g) {
#if USE_ASYNC
  __builtin_amdgcn_global_load_async_to_lds_b128(
      (__attribute__((address_space(1))) i32x4*)g,
      (__attribute__((address_space(3))) i32x4*)l, 0, 0);
#else
  *(u32x4*)l = *(const u32x4*)g;
#endif
}

static __device__ __forceinline__ void wait_async_all() {
#if USE_ASYNC
  __builtin_amdgcn_s_wait_asynccnt(0);
#endif
}

// Load one 16x32 bf16 WMMA operand fragment from LDS.
// Rows are `stride` bf16 apart, K contiguous.
// Lanes 0-15: K 0-7 then 16-23; lanes 16-31: K 8-15 then 24-31.
static __device__ __forceinline__ v16bf load_frag(const unsigned short* base,
                                                  int lane, int stride) {
  const int l16 = lane & 15;
  const int kk0 = (lane & 16) ? 8 : 0;
  const unsigned short* p = base + l16 * stride + kk0;
  union { v16bf v; v8bf h[2]; } u;
  u.h[0] = *(const v8bf*)(p);
  u.h[1] = *(const v8bf*)(p + 16);
  return u.v;
}

// ---------------------------------------------------------------------------
// Pass 0a: x fp32 -> bf16, same layout [E*T, D].
// ---------------------------------------------------------------------------
__global__ __launch_bounds__(256)
void cvt_x_kernel(const float* __restrict__ src, unsigned short* __restrict__ dst) {
  const size_t i4 = ((size_t)blockIdx.x * 256 + threadIdx.x) * 4;
  const float4 f = *(const float4*)(src + i4);
  u32x2 p;
  p[0] = (unsigned)f2bf(f.x) | ((unsigned)f2bf(f.y) << 16);
  p[1] = (unsigned)f2bf(f.z) | ((unsigned)f2bf(f.w) << 16);
  *(u32x2*)(dst + i4) = p;
}

// ---------------------------------------------------------------------------
// Pass 0b: weight fp32 [e][K][N] -> bf16 transposed [e][N][K].
// 64x64 tile per 256-thread block through LDS; coalesced in and out.
// Grid: (N/64, K/64, NE).
// ---------------------------------------------------------------------------
#define TP 72   // padded LDS row stride (bf16 elements)
__global__ __launch_bounds__(256)
void cvt_wT_kernel(const float* __restrict__ w, unsigned short* __restrict__ wt,
                   int K, int N) {
  const int e  = blockIdx.z;
  const int n0 = blockIdx.x * 64;
  const int k0 = blockIdx.y * 64;
  const float* S = w + (size_t)e * K * N;
  unsigned short* Dt = wt + (size_t)e * K * N;

  __shared__ __align__(16) unsigned short Ls[64 * TP]; // [k][n]

  const int tid = threadIdx.x;

  for (int c = tid; c < 1024; c += 256) {
    const int r  = c >> 4;          // k row, 16 float4 per row
    const int cc = (c & 15) << 2;   // n offset
    const float4 f = *(const float4*)(S + (size_t)(k0 + r) * N + n0 + cc);
    u32x2 p;
    p[0] = (unsigned)f2bf(f.x) | ((unsigned)f2bf(f.y) << 16);
    p[1] = (unsigned)f2bf(f.z) | ((unsigned)f2bf(f.w) << 16);
    *(u32x2*)&Ls[r * TP + cc] = p;
  }
  __syncthreads();

  for (int c = tid; c < 512; c += 256) {
    const int rn = c >> 3;          // n row
    const int ck = (c & 7) << 3;    // k offset, 8 bf16 per chunk
    union { u32x4 v; unsigned short s[8]; } u;
#pragma unroll
    for (int j = 0; j < 8; ++j) u.s[j] = Ls[(ck + j) * TP + rn];
    *(u32x4*)(Dt + (size_t)(n0 + rn) * K + k0 + ck) = u.v;
  }
}

// ---------------------------------------------------------------------------
// Staging helpers (one K-stage worth of tiles into one LDS buffer).
// ---------------------------------------------------------------------------
static __device__ __forceinline__ void stage_gateup(
    unsigned short* As, unsigned short* Bs1, unsigned short* Bs3,
    const unsigned short* Ag, const unsigned short* B1g,
    const unsigned short* B3g, int row0, int col0, int k0, int tid) {
  for (int c = tid; c < (BM * BK) / 8; c += NTHREADS) {
    const int r = c >> 3, kk = (c & 7) << 3;
    cp16_g2l(&As[r * BK + kk], Ag + (size_t)(row0 + r) * ND + k0 + kk);
  }
  for (int c = tid; c < (BN * BK) / 8; c += NTHREADS) {
    const int r = c >> 3, kk = (c & 7) << 3;
    const size_t go = (size_t)(col0 + r) * ND + k0 + kk;
    cp16_g2l(&Bs1[r * BK + kk], B1g + go);
    cp16_g2l(&Bs3[r * BK + kk], B3g + go);
  }
}

static __device__ __forceinline__ void stage_down(
    unsigned short* As, unsigned short* Bs,
    const unsigned short* Ag, const unsigned short* Bg,
    int row0, int col0, int k0, int tid) {
  for (int c = tid; c < (BM * BK) / 8; c += NTHREADS) {
    const int r = c >> 3, kk = (c & 7) << 3;
    cp16_g2l(&As[r * BK + kk], Ag + (size_t)(row0 + r) * NH + k0 + kk);
  }
  for (int c = tid; c < (BN * BK) / 8; c += NTHREADS) {
    const int r = c >> 3, kk = (c & 7) << 3;
    cp16_g2l(&Bs[r * BK + kk], Bg + (size_t)(col0 + r) * NH + k0 + kk);
  }
}

// ---------------------------------------------------------------------------
// GEMM 1 (fused): h[e] = silu(x[e] @ w1[e]) * (x[e] @ w3[e]) -> bf16.
// A: xb [e][T][D]; B: w1t/w3t [e][H][D] (pre-transposed, K-contiguous).
// Grid: (T/BM, H/BN, E). Double-buffered LDS, async-DMA pipelined.
// ---------------------------------------------------------------------------
__global__ __launch_bounds__(NTHREADS)
void moe_gateup_kernel(const unsigned short* __restrict__ xb,
                       const unsigned short* __restrict__ w1t,
                       const unsigned short* __restrict__ w3t,
                       unsigned short* __restrict__ hbuf) {
  const int e    = blockIdx.z;
  const int row0 = blockIdx.x * BM;
  const int col0 = blockIdx.y * BN;

  const unsigned short* Ag  = xb  + (size_t)e * NT * ND;
  const unsigned short* B1g = w1t + (size_t)e * NH * ND;
  const unsigned short* B3g = w3t + (size_t)e * NH * ND;
  unsigned short* Hg = hbuf + (size_t)e * NT * NH;

  __shared__ __align__(16) unsigned short As [2 * BM * BK];
  __shared__ __align__(16) unsigned short Bs1[2 * BN * BK];
  __shared__ __align__(16) unsigned short Bs3[2 * BN * BK];

  const int tid  = threadIdx.x;
  const int wave = tid >> 5;
  const int lane = tid & 31;
  const int wm   = (wave >> 2) * 32;  // 0 or 32
  const int wn   = (wave & 3)  * 32;  // 0..96

  v8f accg[2][2], accu[2][2];
  for (int i = 0; i < 2; ++i)
    for (int j = 0; j < 2; ++j) { accg[i][j] = {}; accu[i][j] = {}; }

  stage_gateup(As, Bs1, Bs3, Ag, B1g, B3g, row0, col0, 0, tid);

  const int NIT = ND / BK;
  for (int it = 0; it < NIT; ++it) {
    wait_async_all();
    __syncthreads();

    const int cur = it & 1;
    if (it + 1 < NIT) {
      const int nxt = (it + 1) & 1;
      stage_gateup(As + nxt * BM * BK, Bs1 + nxt * BN * BK, Bs3 + nxt * BN * BK,
                   Ag, B1g, B3g, row0, col0, (it + 1) * BK, tid);
    }

    const unsigned short* Ab  = As  + cur * BM * BK;
    const unsigned short* B1b = Bs1 + cur * BN * BK;
    const unsigned short* B3b = Bs3 + cur * BN * BK;

#pragma unroll
    for (int ks = 0; ks < BK; ks += 32) {
      v16bf af[2], b1f[2], b3f[2];
      af[0]  = load_frag(Ab  + (wm +  0) * BK + ks, lane, BK);
      af[1]  = load_frag(Ab  + (wm + 16) * BK + ks, lane, BK);
      b1f[0] = load_frag(B1b + (wn +  0) * BK + ks, lane, BK);
      b1f[1] = load_frag(B1b + (wn + 16) * BK + ks, lane, BK);
      b3f[0] = load_frag(B3b + (wn +  0) * BK + ks, lane, BK);
      b3f[1] = load_frag(B3b + (wn + 16) * BK + ks, lane, BK);

#pragma unroll
      for (int i = 0; i < 2; ++i)
#pragma unroll
        for (int j = 0; j < 2; ++j) {
          accg[i][j] = __builtin_amdgcn_wmma_f32_16x16x32_bf16(
              false, af[i], false, b1f[j], (short)0, accg[i][j], false, false);
          accu[i][j] = __builtin_amdgcn_wmma_f32_16x16x32_bf16(
              false, af[i], false, b3f[j], (short)0, accu[i][j], false, false);
        }
    }
  }

  // Epilogue: silu(gate) * up -> bf16 h.
  const int l16   = lane & 15;
  const int rhalf = (lane & 16) ? 8 : 0;  // C layout: lanes 16-31 hold M+8
  for (int i = 0; i < 2; ++i)
    for (int j = 0; j < 2; ++j) {
      const int rbase = row0 + wm + i * 16 + rhalf;
      const int cc    = col0 + wn + j * 16 + l16;
#pragma unroll
      for (int r = 0; r < 8; ++r) {
        const float g = accg[i][j][r];
        const float u = accu[i][j][r];
        const float s = g / (1.0f + __expf(-g));
        Hg[(size_t)(rbase + r) * NH + cc] = f2bf(s * u);
      }
    }
}

// ---------------------------------------------------------------------------
// GEMM 2: out[e] = h[e] @ w2[e]   (h bf16 [T][H], w2t bf16 [D][H], out fp32)
// Grid: (T/BM, D/BN, E). Double-buffered LDS, async-DMA pipelined.
// ---------------------------------------------------------------------------
__global__ __launch_bounds__(NTHREADS)
void moe_down_kernel(const unsigned short* __restrict__ hbuf,
                     const unsigned short* __restrict__ w2t,
                     float* __restrict__ out) {
  const int e    = blockIdx.z;
  const int row0 = blockIdx.x * BM;
  const int col0 = blockIdx.y * BN;

  const unsigned short* Ag = hbuf + (size_t)e * NT * NH;
  const unsigned short* Bg = w2t  + (size_t)e * ND * NH;
  float*                Cg = out  + (size_t)e * NT * ND;

  __shared__ __align__(16) unsigned short As[2 * BM * BK];
  __shared__ __align__(16) unsigned short Bs[2 * BN * BK];

  const int tid  = threadIdx.x;
  const int wave = tid >> 5;
  const int lane = tid & 31;
  const int wm   = (wave >> 2) * 32;
  const int wn   = (wave & 3)  * 32;

  v8f acc[2][2];
  for (int i = 0; i < 2; ++i)
    for (int j = 0; j < 2; ++j) acc[i][j] = {};

  stage_down(As, Bs, Ag, Bg, row0, col0, 0, tid);

  const int NIT = NH / BK;
  for (int it = 0; it < NIT; ++it) {
    wait_async_all();
    __syncthreads();

    const int cur = it & 1;
    if (it + 1 < NIT) {
      const int nxt = (it + 1) & 1;
      stage_down(As + nxt * BM * BK, Bs + nxt * BN * BK,
                 Ag, Bg, row0, col0, (it + 1) * BK, tid);
    }

    const unsigned short* Ab = As + cur * BM * BK;
    const unsigned short* Bb = Bs + cur * BN * BK;

#pragma unroll
    for (int ks = 0; ks < BK; ks += 32) {
      v16bf af[2], bf[2];
      af[0] = load_frag(Ab + (wm +  0) * BK + ks, lane, BK);
      af[1] = load_frag(Ab + (wm + 16) * BK + ks, lane, BK);
      bf[0] = load_frag(Bb + (wn +  0) * BK + ks, lane, BK);
      bf[1] = load_frag(Bb + (wn + 16) * BK + ks, lane, BK);

#pragma unroll
      for (int i = 0; i < 2; ++i)
#pragma unroll
        for (int j = 0; j < 2; ++j)
          acc[i][j] = __builtin_amdgcn_wmma_f32_16x16x32_bf16(
              false, af[i], false, bf[j], (short)0, acc[i][j], false, false);
    }
  }

  const int l16   = lane & 15;
  const int rhalf = (lane & 16) ? 8 : 0;
  for (int i = 0; i < 2; ++i)
    for (int j = 0; j < 2; ++j) {
      const int rbase = row0 + wm + i * 16 + rhalf;
      const int cc    = col0 + wn + j * 16 + l16;
#pragma unroll
      for (int r = 0; r < 8; ++r)
        Cg[(size_t)(rbase + r) * ND + cc] = acc[i][j][r];
    }
}

extern "C" void kernel_launch(void* const* d_in, const int* in_sizes, int n_in,
                              void* d_out, int out_size, void* d_ws, size_t ws_size,
                              hipStream_t stream) {
  const float* x  = (const float*)d_in[0];  // [E,T,D]
  const float* w1 = (const float*)d_in[1];  // [E,D,H]
  const float* w2 = (const float*)d_in[2];  // [E,H,D]
  const float* w3 = (const float*)d_in[3];  // [E,D,H]
  float* out = (float*)d_out;               // [E,T,D]

  // Workspace layout (bf16 elements): xb | w1t | w3t | w2t | hb  (~151 MB)
  unsigned short* ws = (unsigned short*)d_ws;
  size_t off = 0;
  unsigned short* xb  = ws + off; off += (size_t)NE * NT * ND;   // [E,T,D]
  unsigned short* w1t = ws + off; off += (size_t)NE * NH * ND;   // [E,H,D]
  unsigned short* w3t = ws + off; off += (size_t)NE * NH * ND;   // [E,H,D]
  unsigned short* w2t = ws + off; off += (size_t)NE * ND * NH;   // [E,D,H]
  unsigned short* hb  = ws + off;                                 // [E,T,H]

  dim3 blk(NTHREADS);

  // Pass 0: downcast + weight transpose.
  cvt_x_kernel<<<dim3(((size_t)NE * NT * ND) / 4 / 256), dim3(256), 0, stream>>>(x, xb);
  cvt_wT_kernel<<<dim3(NH / 64, ND / 64, NE), dim3(256), 0, stream>>>(w1, w1t, ND, NH);
  cvt_wT_kernel<<<dim3(NH / 64, ND / 64, NE), dim3(256), 0, stream>>>(w3, w3t, ND, NH);
  cvt_wT_kernel<<<dim3(ND / 64, NH / 64, NE), dim3(256), 0, stream>>>(w2, w2t, NH, ND);

  // GEMM 1: fused gate/up + SwiGLU.
  moe_gateup_kernel<<<dim3(NT / BM, NH / BN, NE), blk, 0, stream>>>(xb, w1t, w3t, hb);
  // GEMM 2: down projection.
  moe_down_kernel<<<dim3(NT / BM, ND / BN, NE), blk, 0, stream>>>(hb, w2t, out);
}
